// selfalignment_53163105190092
// MI455X (gfx1250) — matrix operand order repaired and checked by reading, not compiled, not run on hardware
//
#include <hip/hip_runtime.h>

#define S1 1024
#define S2 1024
#define DD 1024
#define NBATCH 16

typedef __attribute__((ext_vector_type(16))) __bf16 v16bf;
typedef __attribute__((ext_vector_type(8)))  float  v8f;

__device__ __forceinline__ v8f wmma_bf16(v16bf a, v16bf b, v8f c) {
  return __builtin_amdgcn_wmma_f32_16x16x32_bf16(false, a, false, b, (short)0, c, false, false);
}

// ---------- fragment loaders (contiguous k order, wave32) ----------

// A 16x32 bf16, row-major in memory, K contiguous. lane&15 = M row, lane>>4 selects K-halves.
// half h -> K = h + 16*g  => one contiguous 16-element (32B) run per lane.
__device__ __forceinline__ v16bf load_a_frag(const __bf16* __restrict__ p, int lda) {
  int lane = threadIdx.x & 31;
  int m = lane & 15, g = lane >> 4;
  const __bf16* r = p + (size_t)m * lda + 16 * g;
  v16bf a;
#pragma unroll
  for (int i = 0; i < 8; ++i) a[i] = r[i];
#pragma unroll
  for (int i = 0; i < 8; ++i) a[8 + i] = r[8 + i];
  return a;
}

// B 32x16 bf16 where B[k][n] = Y[n][k] (i.e. computing X @ Y^T with Y row-major, K contiguous).
// half h -> K = h + 16*g, N = lane&15 => one contiguous 16-element (32B) run per lane.
__device__ __forceinline__ v16bf load_b_frag_T(const __bf16* __restrict__ y, int ldy) {
  int lane = threadIdx.x & 31;
  int n = lane & 15, g = lane >> 4;
  const __bf16* r = y + (size_t)n * ldy + 16 * g;
  v16bf b;
#pragma unroll
  for (int i = 0; i < 16; ++i) b[i] = r[i];
  return b;
}

// ---------- 32x32 wave-tile GEMM core: C(32x32) += A(32xK) * Y(32xK)^T ----------

__device__ __forceinline__ void mm_abT(const __bf16* __restrict__ A, int lda,
                                       const __bf16* __restrict__ Y, int ldy,
                                       int K, v8f acc[2][2]) {
  for (int k = 0; k < K; k += 32) {
    v16bf a0 = load_a_frag(A + k, lda);
    v16bf a1 = load_a_frag(A + (size_t)16 * lda + k, lda);
    v16bf b0 = load_b_frag_T(Y + k, ldy);
    v16bf b1 = load_b_frag_T(Y + (size_t)16 * ldy + k, ldy);
    acc[0][0] = wmma_bf16(a0, b0, acc[0][0]);
    acc[0][1] = wmma_bf16(a0, b1, acc[0][1]);
    acc[1][0] = wmma_bf16(a1, b0, acc[1][0]);
    acc[1][1] = wmma_bf16(a1, b1, acc[1][1]);
  }
}

// ---------- kernels ----------

__global__ void k_cvt(const float* __restrict__ x, __bf16* __restrict__ y, int n) {
  int i = blockIdx.x * blockDim.x + threadIdx.x;
  if (i < n) y[i] = (__bf16)x[i];
}

// 32x32 LDS tile transpose + f32->bf16 convert: x[b, r, c] -> xt[b, c, r]  (1024x1024 per batch)
__global__ void k_transpose_cvt(const float* __restrict__ x, __bf16* __restrict__ xt) {
  __shared__ __bf16 tile[32][33];
  int bid = blockIdx.x;
  int b  = bid >> 10;            // 32*32 tiles per batch
  int tr = (bid >> 5) & 31;
  int tc = bid & 31;
  int tx = threadIdx.x & 31, ty = threadIdx.x >> 5;   // 32x8 threads
  const float* xb = x + (size_t)b * 1024 * 1024;
  __bf16* xtb = xt + (size_t)b * 1024 * 1024;
  int r0 = tr * 32, c0 = tc * 32;
#pragma unroll
  for (int i = 0; i < 4; ++i)
    tile[ty + 8 * i][tx] = (__bf16)xb[(size_t)(r0 + ty + 8 * i) * 1024 + c0 + tx];
  __syncthreads();
#pragma unroll
  for (int i = 0; i < 4; ++i)
    xtb[(size_t)(c0 + ty + 8 * i) * 1024 + r0 + tx] = tile[tx][ty + 8 * i];
}

// proj[b,s,d] = sum_k text[b,s,k] * W[d,k]   (bf16 out)
__global__ void k_gemm_proj(const __bf16* __restrict__ tA, const __bf16* __restrict__ wbf,
                            __bf16* __restrict__ projbf) {
  int wid = (blockIdx.x * blockDim.x + threadIdx.x) >> 5;
  int b = wid >> 10, mt = (wid >> 5) & 31, nt = wid & 31;
  const __bf16* A = tA + (size_t)b * S1 * DD + (size_t)(mt * 32) * DD;
  const __bf16* Y = wbf + (size_t)(nt * 32) * DD;
  v8f acc[2][2] = {};
  mm_abT(A, DD, Y, DD, DD, acc);
  __bf16* C = projbf + (size_t)b * S1 * DD + (size_t)(mt * 32) * DD + nt * 32;
  int lane = threadIdx.x & 31;
  int n = lane & 15, g = lane >> 4;
#pragma unroll
  for (int i = 0; i < 2; ++i)
#pragma unroll
    for (int j = 0; j < 2; ++j)
#pragma unroll
      for (int r = 0; r < 8; ++r)
        C[(size_t)(i * 16 + r + 8 * g) * DD + j * 16 + n] = (__bf16)acc[i][j][r];
}

// masked[b,s,t] = proj[b,s,:] . text1[b,t,:] + (1 - m1[b,s]*m2[b,t]) * -10000
__global__ void k_gemm_logit(const __bf16* __restrict__ projbf, const __bf16* __restrict__ tB,
                             const float* __restrict__ m1, const float* __restrict__ m2,
                             float* __restrict__ masked) {
  int wid = (blockIdx.x * blockDim.x + threadIdx.x) >> 5;
  int b = wid >> 10, mt = (wid >> 5) & 31, nt = wid & 31;
  const __bf16* A = projbf + (size_t)b * S1 * DD + (size_t)(mt * 32) * DD;
  const __bf16* Y = tB + (size_t)b * S2 * DD + (size_t)(nt * 32) * DD;
  v8f acc[2][2] = {};
  mm_abT(A, DD, Y, DD, DD, acc);
  float* M = masked + (size_t)b * S1 * S2;
  const float* m1b = m1 + b * S1;
  const float* m2b = m2 + b * S2;
  int lane = threadIdx.x & 31;
  int n = lane & 15, g = lane >> 4;
#pragma unroll
  for (int i = 0; i < 2; ++i)
#pragma unroll
    for (int r = 0; r < 8; ++r) {
      int s = mt * 32 + i * 16 + r + 8 * g;
      float a1v = m1b[s];
#pragma unroll
      for (int j = 0; j < 2; ++j) {
        int t = nt * 32 + j * 16 + n;
        M[(size_t)s * S2 + t] = acc[i][j][r] + (1.0f - a1v * m2b[t]) * -10000.0f;
      }
    }
}

// softmax over last axis (S2); writes bf16 probs and logits*mask1 (f32) output
__global__ void k_softmax_rows(const float* __restrict__ masked, const float* __restrict__ m1,
                               __bf16* __restrict__ lgbf, float* __restrict__ out3) {
  __shared__ float red[8];
  int row = blockIdx.x;                       // b*S1 + s
  const float* x = masked + (size_t)row * S2;
  int tid = threadIdx.x;
  float v[4];
  float mx = -3.0e38f;
#pragma unroll
  for (int i = 0; i < 4; ++i) { v[i] = x[tid + 256 * i]; mx = fmaxf(mx, v[i]); }
#pragma unroll
  for (int o = 16; o > 0; o >>= 1) mx = fmaxf(mx, __shfl_xor(mx, o, 32));
  if ((tid & 31) == 0) red[tid >> 5] = mx;
  __syncthreads();
  mx = red[0];
#pragma unroll
  for (int i = 1; i < 8; ++i) mx = fmaxf(mx, red[i]);
  __syncthreads();
  float s = 0.0f;
#pragma unroll
  for (int i = 0; i < 4; ++i) { v[i] = __expf(v[i] - mx); s += v[i]; }
#pragma unroll
  for (int o = 16; o > 0; o >>= 1) s += __shfl_xor(s, o, 32);
  if ((tid & 31) == 0) red[tid >> 5] = s;
  __syncthreads();
  s = 0.0f;
#pragma unroll
  for (int i = 0; i < 8; ++i) s += red[i];
  float inv = 1.0f / s;
  float mv = m1[row];
#pragma unroll
  for (int i = 0; i < 4; ++i) {
    float p = v[i] * inv;
    lgbf[(size_t)row * S2 + tid + 256 * i] = (__bf16)p;
    out3[(size_t)row * S2 + tid + 256 * i] = p * mv;
  }
}

// softmax over axis -2 (S1), one thread per column t; writes TRANSPOSED bf16: lg1T[b, t, s]
__global__ void k_softmax_cols(const float* __restrict__ masked, __bf16* __restrict__ lg1T) {
  int col = blockIdx.x * blockDim.x + threadIdx.x;   // b*S2 + t
  int b = col >> 10, t = col & 1023;
  const float* base = masked + (size_t)b * S1 * S2 + t;
  float mx = -3.0e38f;
  for (int sI = 0; sI < S1; ++sI) mx = fmaxf(mx, base[(size_t)sI * S2]);
  float sum = 0.0f;
  for (int sI = 0; sI < S1; ++sI) sum += __expf(base[(size_t)sI * S2] - mx);
  float inv = 1.0f / sum;
  __bf16* o = lg1T + (size_t)b * S2 * S1 + (size_t)t * S1;   // row t, s contiguous
  for (int sI = 0; sI < S1; ++sI)
    o[sI] = (__bf16)(__expf(base[(size_t)sI * S2] - mx) * inv);
}

// out[b,s,d] = (sum_t p[b,s,t]*text1[b,t,d]) * m1[b,s] + text[b,s,d]
// Y = tBT[b, d, t] (text1 transposed, t contiguous) -> clean X @ Y^T path
__global__ void k_gemm_out(const __bf16* __restrict__ lgbf, const __bf16* __restrict__ tBT,
                           const float* __restrict__ m1, const float* __restrict__ text,
                           float* __restrict__ out) {
  int wid = (blockIdx.x * blockDim.x + threadIdx.x) >> 5;
  int b = wid >> 10, mt = (wid >> 5) & 31, nt = wid & 31;
  const __bf16* A = lgbf + (size_t)b * S1 * S2 + (size_t)(mt * 32) * S2;
  const __bf16* Y = tBT + (size_t)b * DD * S2 + (size_t)(nt * 32) * S2;
  v8f acc[2][2] = {};
  mm_abT(A, S2, Y, S2, S2, acc);
  const float* m1b = m1 + b * S1;
  int lane = threadIdx.x & 31;
  int n = lane & 15, g = lane >> 4;
#pragma unroll
  for (int i = 0; i < 2; ++i)
#pragma unroll
    for (int r = 0; r < 8; ++r) {
      int s = mt * 32 + i * 16 + r + 8 * g;
      float mv = m1b[s];
#pragma unroll
      for (int j = 0; j < 2; ++j) {
        int d = nt * 32 + j * 16 + n;
        size_t idx = (size_t)b * S1 * DD + (size_t)s * DD + d;
        out[idx] = acc[i][j][r] * mv + text[idx];
      }
    }
}

// out1[b,t,d] = (sum_s p1T[b,t,s]*text[b,s,d]) * m2[b,t] + text1[b,t,d]
// A = lg1T[b, t, s] (s contiguous), Y = tAT[b, d, s] (text transposed, s contiguous)
__global__ void k_gemm_out1(const __bf16* __restrict__ lg1T, const __bf16* __restrict__ tAT,
                            const float* __restrict__ m2, const float* __restrict__ text1,
                            float* __restrict__ out1) {
  int wid = (blockIdx.x * blockDim.x + threadIdx.x) >> 5;
  int b = wid >> 10, mt = (wid >> 5) & 31, nt = wid & 31;
  const __bf16* A = lg1T + (size_t)b * S2 * S1 + (size_t)(mt * 32) * S1;
  const __bf16* Y = tAT + (size_t)b * DD * S1 + (size_t)(nt * 32) * S1;
  v8f acc[2][2] = {};
  mm_abT(A, S1, Y, S1, S1, acc);
  const float* m2b = m2 + b * S2;
  int lane = threadIdx.x & 31;
  int n = lane & 15, g = lane >> 4;
#pragma unroll
  for (int i = 0; i < 2; ++i)
#pragma unroll
    for (int r = 0; r < 8; ++r) {
      int t = mt * 32 + i * 16 + r + 8 * g;
      float mv = m2b[t];
#pragma unroll
      for (int j = 0; j < 2; ++j) {
        int d = nt * 32 + j * 16 + n;
        size_t idx = (size_t)b * S2 * DD + (size_t)t * DD + d;
        out1[idx] = acc[i][j][r] * mv + text1[idx];
      }
    }
}

// ---------- host launcher ----------

extern "C" void kernel_launch(void* const* d_in, const int* in_sizes, int n_in,
                              void* d_out, int out_size, void* d_ws, size_t ws_size,
                              hipStream_t stream) {
  const float* text  = (const float*)d_in[0];   // [B,S1,D]
  const float* text1 = (const float*)d_in[1];   // [B,S2,D]
  const float* m1    = (const float*)d_in[2];   // [B,S1]
  const float* m2    = (const float*)d_in[3];   // [B,S2]
  const float* W     = (const float*)d_in[4];   // [D,D]

  float* out0 = (float*)d_out;                           // output  [B,S1,D]
  float* out1 = out0 + (size_t)NBATCH * S1 * DD;         // output1 [B,S2,D]
  float* out3 = out1 + (size_t)NBATCH * S2 * DD;         // logits*mask1 [B,S1,S2]

  char* ws = (char*)d_ws;
  __bf16* tA     = (__bf16*)(ws);                        //  32 MB text bf16
  __bf16* tB     = (__bf16*)(ws + 33554432ull);          //  32 MB text1 bf16
  __bf16* wbf    = (__bf16*)(ws + 67108864ull);          //   2 MB W bf16
  __bf16* projbf = (__bf16*)(ws + 69206016ull);          //  32 MB proj bf16
  float*  masked = (float*) (ws + 102760448ull);         //  64 MB masked logit f32
  __bf16* lg1T   = (__bf16*)(ws + 169869312ull);         //  32 MB softmax-S1 bf16 (transposed)
  __bf16* lgbf   = projbf;                               //  alias: proj dead after gemm_logit
  // masked region is dead after both softmaxes -> reuse for the transposed bf16 copies:
  __bf16* tBT    = (__bf16*)(ws + 102760448ull);         //  32 MB text1^T bf16 [B,D,S2]
  __bf16* tAT    = (__bf16*)(ws + 136314880ull);         //  32 MB text^T  bf16 [B,D,S1]

  const int nTxt = NBATCH * S1 * DD;   // 16777216
  const int nW   = DD * DD;            // 1048576

  k_cvt<<<nTxt / 256, 256, 0, stream>>>(text,  tA,  nTxt);
  k_cvt<<<nTxt / 256, 256, 0, stream>>>(text1, tB,  nTxt);
  k_cvt<<<nW   / 256, 256, 0, stream>>>(W,     wbf, nW);

  // 16 batches * 32x32 tiles of 32 => 1024 waves/batch, 8 waves/block
  const int gemmBlocks = NBATCH * 32 * 32 / 8;   // 2048
  const int trBlocks   = NBATCH * 32 * 32;       // 16384 (32x32 tiles)

  k_gemm_proj <<<gemmBlocks, 256, 0, stream>>>(tA, wbf, projbf);
  k_gemm_logit<<<gemmBlocks, 256, 0, stream>>>(projbf, tB, m1, m2, masked);

  k_softmax_rows<<<NBATCH * S1, 256, 0, stream>>>(masked, m1, lgbf, out3);
  k_softmax_cols<<<NBATCH * S2 / 256, 256, 0, stream>>>(masked, lg1T);

  // masked is dead now; build transposed bf16 operands in its place
  k_transpose_cvt<<<trBlocks, 256, 0, stream>>>(text1, tBT);
  k_transpose_cvt<<<trBlocks, 256, 0, stream>>>(text,  tAT);

  k_gemm_out <<<gemmBlocks, 256, 0, stream>>>(lgbf, tBT, m1, text,  out0);
  k_gemm_out1<<<gemmBlocks, 256, 0, stream>>>(lg1T, tAT, m2, text1, out1);
}
